// QuaternionLSTM_22273700397455
// MI455X (gfx1250) — matrix operand order repaired and checked by reading, not compiled, not run on hardware
//
#include <hip/hip_runtime.h>

// ============================================================================
// Quaternion LSTM (single step, h0 = c0 = 0) for MI455X / gfx1250.
//
// Math reduction: h0 == 0  =>  gu == 0 and ft*c0 == 0, so the forget gate and
// all uh_* weights are dead. Computation is:
//   z[g]  = x @ Kx[g] + bx[g]      for g in {1,2,3}   (input, output, cell)
//   h     = sigmoid(z2) * tanh( sigmoid(z1) * tanh(z3) )
//   out   = h @ fco_w + fco_b
//
// Roofline: ~56 GFLOP vs ~178 MB mandatory HBM traffic => memory-bound at
// 23.3 TB/s (~7.6us floor). bf16 WMMA (v_wmma_f32_16x16x32_bf16) keeps the
// compute far below that; x is converted fp32->bf16 on the fly through LDS
// (v_cvt_pk_bf16_f32) so it leaves HBM exactly once.
//
// Weight tiles move global->LDS via CDNA5 async DMA
// (global_load_async_to_lds_b128 / ASYNCcnt, probe-confirmed signature:
// (as1 v4i*, as3 v4i*, imm offset, imm cpol)), issued for the next LDS buffer
// before the WMMA chain and waited with s_wait_asynccnt after it.
// B fragments stream through an explicit 3-deep rotating queue so each WMMA
// needs only a partial s_wait_dscnt instead of a full drain.
// ============================================================================

typedef __attribute__((ext_vector_type(16))) __bf16 v16bf;
typedef __attribute__((ext_vector_type(2)))  __bf16 v2bf;
typedef __attribute__((ext_vector_type(8)))  float  v8f;

union FragBF { uint4 u[2]; v16bf v; };

#if __has_builtin(__builtin_amdgcn_global_load_async_to_lds_b128)
#define HAVE_ASYNC_LDS 1
#else
#define HAVE_ASYNC_LDS 0
#endif

#if HAVE_ASYNC_LDS
// b128 payload type: GCC-style int vector of 4 (matches clang's V4i proto)
typedef int v4i_vs __attribute__((vector_size(16)));
typedef __attribute__((address_space(1))) v4i_vs as1_v4i;
typedef __attribute__((address_space(3))) v4i_vs as3_v4i;

__device__ __forceinline__ void async_b128(const unsigned short* g,
                                           unsigned short* l) {
  __builtin_amdgcn_global_load_async_to_lds_b128(
      (as1_v4i*)(void*)const_cast<unsigned short*>(g),
      (as3_v4i*)(void*)l, 0, 0);
}
#endif

__device__ __forceinline__ void wait_async0() {
#if HAVE_ASYNC_LDS
#if __has_builtin(__builtin_amdgcn_s_wait_asynccnt)
  __builtin_amdgcn_s_wait_asynccnt(0);
#else
  asm volatile("s_wait_asynccnt 0x0" ::: "memory");
#endif
#endif
}

__device__ __forceinline__ unsigned short f2bf(float f) {
  __bf16 b = (__bf16)f;                 // selects v_cvt bf16 hw op
  return __builtin_bit_cast(unsigned short, b);
}

__device__ __forceinline__ unsigned int pack_bf2(float lo, float hi) {
  v2bf v;                               // selects v_cvt_pk_bf16_f32
  v.x = (__bf16)lo;
  v.y = (__bf16)hi;
  return __builtin_bit_cast(unsigned int, v);
}

__device__ __forceinline__ float fast_sigmoid(float x) {
  return 1.0f / (1.0f + __expf(-x));
}
__device__ __forceinline__ float fast_tanh(float x) {
  float e = __expf(2.0f * x);
  return 1.0f - 2.0f / (e + 1.0f);
}

// LDS row stride in bf16 elements. 40 elems = 80B: fragment b128 reads by 16
// consecutive lanes land on banks 20*l..20*l+3 (mod 64) -> conflict-free.
#define LSTR 40

// ----------------------------------------------------------------------------
// Prep A: build transposed quaternion block kernel Wt[n][k] (bf16).
// ----------------------------------------------------------------------------
__global__ __launch_bounds__(256) void build_wt_kernel(
    const float* __restrict__ wr, const float* __restrict__ wi,
    const float* __restrict__ wj, const float* __restrict__ wk,
    unsigned short* __restrict__ Wt) {
  int idx = blockIdx.x * 256 + threadIdx.x;   // 1024*1024 elements
  int n = idx >> 10, k = idx & 1023;
  int g  = n >> 8,  hc = n & 255;
  int cb = hc >> 6, q  = hc & 63;
  int rb = k >> 8,  p  = k & 255;
  const int   comp_tab[16] = {0,1,2,3,  1,0,3,2,  2,3,0,1,  3,2,1,0};
  const float sgn_tab[16]  = {1,1,1,1, -1,1,1,-1, -1,-1,1,1, -1,1,-1,1};
  int sel = rb * 4 + cb;
  int c   = comp_tab[sel];
  const float* base = (c == 0) ? wr : (c == 1) ? wi : (c == 2) ? wj : wk;
  float v = sgn_tab[sel] * base[(g * 256 + p) * 64 + q];
  Wt[(size_t)n * 1024 + k] = f2bf(v);
}

// Prep B: Wt2[n][k] = fco_w[k][n] in bf16 (column-major for B fragments)
__global__ __launch_bounds__(256) void build_w2_kernel(
    const float* __restrict__ fco_w, unsigned short* __restrict__ Wt2) {
  int idx = blockIdx.x * 256 + threadIdx.x;   // 256*256 elements
  int n = idx >> 8, k = idx & 255;
  Wt2[n * 256 + k] = f2bf(fco_w[k * 256 + n]);
}

// ----------------------------------------------------------------------------
// GEMM1 + fused activation: 128-row x (3 gates x 64 col) tile per workgroup.
// ----------------------------------------------------------------------------
__global__ __launch_bounds__(256) void qlstm_gate_kernel(
    const float* __restrict__ x, const unsigned short* __restrict__ Wt,
    const float* __restrict__ bx, unsigned short* __restrict__ hbuf) {
  __shared__ unsigned short lA[2][128 * LSTR];   // x tile, bf16
  __shared__ unsigned short lB[2][192 * LSTR];   // weight tile, bf16 col-major

  const int tid    = threadIdx.x;
  const int wave   = tid >> 5;
  const int lane   = tid & 31;
  const int laneN  = lane & 15;
  const int hiHalf = lane >> 4;
  const int m0     = blockIdx.x * 128;
  const int cg     = blockIdx.y;              // h-column group [cg*64, +64)

  v8f acc[12] = {};                           // [gate(3) x colTile(4)]

  const int  xrow  = tid >> 1;                // 128 rows, 2 threads/row
  const int  xkh   = (tid & 1) * 16;
  const bool wldEn = (tid < 192);
  const int  gi    = tid >> 6;                // 0..2 -> actual gate gi+1
  const int  goff  = tid & 63;

  const float* xbase = x + (size_t)(m0 + xrow) * 1024 + xkh;
  const unsigned short* wbase =
      Wt + (size_t)((gi + 1) * 256 + cg * 64 + goff) * 1024;

  float4 xr[4];   // in-flight x tile (fp32) -- the only register staging

  auto load_x = [&](int kb) {
    const float4* xs = reinterpret_cast<const float4*>(xbase + kb * 32);
    xr[0] = xs[0]; xr[1] = xs[1]; xr[2] = xs[2]; xr[3] = xs[3];
  };
  auto store_x = [&](int buf) {
    uint4 p0, p1;
    p0.x = pack_bf2(xr[0].x, xr[0].y); p0.y = pack_bf2(xr[0].z, xr[0].w);
    p0.z = pack_bf2(xr[1].x, xr[1].y); p0.w = pack_bf2(xr[1].z, xr[1].w);
    p1.x = pack_bf2(xr[2].x, xr[2].y); p1.y = pack_bf2(xr[2].z, xr[2].w);
    p1.z = pack_bf2(xr[3].x, xr[3].y); p1.w = pack_bf2(xr[3].z, xr[3].w);
    uint4* da = reinterpret_cast<uint4*>(&lA[buf][xrow * LSTR + xkh]);
    da[0] = p0; da[1] = p1;
  };
  // Weight tile: global -> LDS with NO VGPR staging (async DMA).
  auto copy_w = [&](int buf, int kb) {
    if (wldEn) {
      const unsigned short* ws = wbase + kb * 32;
      unsigned short* ld = &lB[buf][tid * LSTR];
#if HAVE_ASYNC_LDS
      async_b128(ws, ld); async_b128(ws + 8, ld + 8);
      async_b128(ws + 16, ld + 16); async_b128(ws + 24, ld + 24);
#else
      const uint4* s4 = reinterpret_cast<const uint4*>(ws);
      uint4 w0 = s4[0], w1 = s4[1], w2 = s4[2], w3 = s4[3];
      uint4* d4 = reinterpret_cast<uint4*>(ld);
      d4[0] = w0; d4[1] = w1; d4[2] = w2; d4[3] = w3;
#endif
    }
  };

  load_x(0);
  copy_w(0, 0);
  store_x(0);
  wait_async0();
  __syncthreads();

  for (int kb = 0; kb < 32; ++kb) {           // K = 1024 = 32 x 32
    const int cur = kb & 1;
    if (kb + 1 < 32) {
      load_x(kb + 1);                         // VMEM issued before the chain
#if HAVE_ASYNC_LDS
      copy_w(1 - cur, kb + 1);                // DMA rides under the WMMAs
#endif
    }

    // A fragment (16x32 bf16, wave32 layout)
    FragBF a;
    {
      const uint4* ap =
          reinterpret_cast<const uint4*>(&lA[cur][(wave * 16 + laneN) * LSTR]);
      a.u[0] = ap[2 * hiHalf];
      a.u[1] = ap[2 * hiHalf + 1];
    }

    // B fragments: explicit 3-deep rotating queue -> partial dscnt waits.
    FragBF bq[3];
    auto ldb = [&](FragBF& f, int t) {
      const uint4* bp =
          reinterpret_cast<const uint4*>(&lB[cur][(t * 16 + laneN) * LSTR]);
      f.u[0] = bp[2 * hiHalf];
      f.u[1] = bp[2 * hiHalf + 1];
    };
    ldb(bq[0], 0);
    ldb(bq[1], 1);
#pragma unroll
    for (int t = 0; t < 12; ++t) {
      if (t + 2 < 12) ldb(bq[(t + 2) % 3], t + 2);
      acc[t] = __builtin_amdgcn_wmma_f32_16x16x32_bf16(
          false, a.v, false, bq[t % 3].v, (short)0, acc[t], false, false);
    }

    if (kb + 1 < 32) {
      store_x(1 - cur);
#if !HAVE_ASYNC_LDS
      copy_w(1 - cur, kb + 1);                // short live range: load+store
#endif
    }
    wait_async0();
    __syncthreads();
  }

  // Fused epilogue: bias + LSTM activations, write h (bf16).
#pragma unroll
  for (int j = 0; j < 4; ++j) {
    int col = cg * 64 + j * 16 + laneN;        // h column in [0,256)
    float bi = bx[1 * 256 + col];
    float bo = bx[2 * 256 + col];
    float ba = bx[3 * 256 + col];
#pragma unroll
    for (int e = 0; e < 8; ++e) {
      int mrow = m0 + wave * 16 + e + hiHalf * 8;
      float zi = acc[0 * 4 + j][e] + bi;       // input gate
      float zo = acc[1 * 4 + j][e] + bo;       // output gate
      float za = acc[2 * 4 + j][e] + ba;       // cell candidate
      float ct = fast_sigmoid(zi) * fast_tanh(za);
      float hv = fast_sigmoid(zo) * fast_tanh(ct);
      hbuf[(size_t)mrow * 256 + col] = f2bf(hv);
    }
  }
}

// ----------------------------------------------------------------------------
// GEMM2: out[B,256] = h @ fco_w + fco_b. Both tiles are pure bf16 copies:
// async path has ZERO register staging.
// ----------------------------------------------------------------------------
__global__ __launch_bounds__(256) void fco_gemm_kernel(
    const unsigned short* __restrict__ hbuf,
    const unsigned short* __restrict__ Wt2,
    const float* __restrict__ fco_b, float* __restrict__ out) {
  __shared__ unsigned short lA[2][128 * LSTR];
  __shared__ unsigned short lB[2][256 * LSTR];

  const int tid    = threadIdx.x;
  const int wave   = tid >> 5;
  const int lane   = tid & 31;
  const int laneN  = lane & 15;
  const int hiHalf = lane >> 4;
  const int m0     = blockIdx.x * 128;

  v8f acc[16] = {};

  const int arow = tid >> 1;
  const int akh  = (tid & 1) * 16;

  const unsigned short* abase = hbuf + (size_t)(m0 + arow) * 256 + akh;
  const unsigned short* wbase = Wt2 + (size_t)tid * 256;

#if HAVE_ASYNC_LDS
  auto copy_tiles = [&](int buf, int kb) {
    const unsigned short* hs = abase + kb * 32;
    unsigned short* la = &lA[buf][arow * LSTR + akh];
    async_b128(hs, la); async_b128(hs + 8, la + 8);
    const unsigned short* ws = wbase + kb * 32;
    unsigned short* lb = &lB[buf][tid * LSTR];
    async_b128(ws, lb); async_b128(ws + 8, lb + 8);
    async_b128(ws + 16, lb + 16); async_b128(ws + 24, lb + 24);
  };
#else
  uint4 hr[2];
  uint4 wq[4];
  auto load_tile = [&](int kb) {
    const uint4* hs = reinterpret_cast<const uint4*>(abase + kb * 32);
    hr[0] = hs[0]; hr[1] = hs[1];
    const uint4* ws = reinterpret_cast<const uint4*>(wbase + kb * 32);
    wq[0] = ws[0]; wq[1] = ws[1]; wq[2] = ws[2]; wq[3] = ws[3];
  };
  auto store_tile = [&](int buf) {
    uint4* da = reinterpret_cast<uint4*>(&lA[buf][arow * LSTR + akh]);
    da[0] = hr[0]; da[1] = hr[1];
    uint4* db = reinterpret_cast<uint4*>(&lB[buf][tid * LSTR]);
    db[0] = wq[0]; db[1] = wq[1]; db[2] = wq[2]; db[3] = wq[3];
  };
#endif

#if HAVE_ASYNC_LDS
  copy_tiles(0, 0);
  wait_async0();
#else
  load_tile(0);
  store_tile(0);
#endif
  __syncthreads();

  for (int kb = 0; kb < 8; ++kb) {            // K = 256 = 8 x 32
    const int cur = kb & 1;
#if HAVE_ASYNC_LDS
    if (kb + 1 < 8) copy_tiles(1 - cur, kb + 1);
#else
    if (kb + 1 < 8) load_tile(kb + 1);
#endif

    FragBF a;
    {
      const uint4* ap =
          reinterpret_cast<const uint4*>(&lA[cur][(wave * 16 + laneN) * LSTR]);
      a.u[0] = ap[2 * hiHalf];
      a.u[1] = ap[2 * hiHalf + 1];
    }

    FragBF bq[3];
    auto ldb = [&](FragBF& f, int t) {
      const uint4* bp =
          reinterpret_cast<const uint4*>(&lB[cur][(t * 16 + laneN) * LSTR]);
      f.u[0] = bp[2 * hiHalf];
      f.u[1] = bp[2 * hiHalf + 1];
    };
    ldb(bq[0], 0);
    ldb(bq[1], 1);
#pragma unroll
    for (int t = 0; t < 16; ++t) {
      if (t + 2 < 16) ldb(bq[(t + 2) % 3], t + 2);
      acc[t] = __builtin_amdgcn_wmma_f32_16x16x32_bf16(
          false, a.v, false, bq[t % 3].v, (short)0, acc[t], false, false);
    }

#if !HAVE_ASYNC_LDS
    if (kb + 1 < 8) store_tile(1 - cur);
#endif
    wait_async0();
    __syncthreads();
  }

#pragma unroll
  for (int t = 0; t < 16; ++t) {
    int col = t * 16 + laneN;
    float bb = fco_b[col];
#pragma unroll
    for (int e = 0; e < 8; ++e) {
      int mrow = m0 + wave * 16 + e + hiHalf * 8;
      out[(size_t)mrow * 256 + col] = acc[t][e] + bb;
    }
  }
}

// ----------------------------------------------------------------------------
// Inputs (setup_inputs order):
//  0:x [32768,1024] f32   1-4: wx_r/i/j/k [4,256,64] f32   5: bx [4,256] f32
//  6-9: uh_* (DEAD: h0==0)   10: fco_w [256,256] f32   11: fco_b [256] f32
// Workspace: Wt (2MB bf16) | Wt2 (128KB bf16) | h (16MB bf16)  ~= 18.2 MB
// ----------------------------------------------------------------------------
extern "C" void kernel_launch(void* const* d_in, const int* in_sizes, int n_in,
                              void* d_out, int out_size, void* d_ws, size_t ws_size,
                              hipStream_t stream) {
  (void)in_sizes; (void)n_in; (void)out_size; (void)ws_size;
  const float* x   = (const float*)d_in[0];
  const float* wr  = (const float*)d_in[1];
  const float* wi  = (const float*)d_in[2];
  const float* wj  = (const float*)d_in[3];
  const float* wk  = (const float*)d_in[4];
  const float* bx  = (const float*)d_in[5];
  const float* fcw = (const float*)d_in[10];
  const float* fcb = (const float*)d_in[11];
  float* out = (float*)d_out;

  unsigned short* Wt  = (unsigned short*)d_ws;        // [1024][1024]
  unsigned short* Wt2 = Wt + 1024 * 1024;             // [256][256]
  unsigned short* hb  = Wt2 + 256 * 256;              // [32768][256]

  build_wt_kernel<<<4096, 256, 0, stream>>>(wr, wi, wj, wk, Wt);
  build_w2_kernel<<<256, 256, 0, stream>>>(fcw, Wt2);
  qlstm_gate_kernel<<<dim3(256, 4), 256, 0, stream>>>(x, Wt, bx, hb);
  fco_gemm_kernel<<<256, 256, 0, stream>>>(hb, Wt2, fcb, out);
}
